// SelfAttention_8727373545516
// MI455X (gfx1250) — matrix operand order repaired and checked by reading, not compiled, not run on hardware
//
#include <hip/hip_runtime.h>
#include <hip/hip_bf16.h>
#include <stdint.h>

typedef __attribute__((ext_vector_type(16))) _Float16 v16h;
typedef __attribute__((ext_vector_type(8)))  float    v8f;
typedef __attribute__((ext_vector_type(2)))  _Float16 h2v;

#define BATCH    2
#define SEQ      4096
#define D_MODEL  512
#define NUM_H    8
#define DIM_H    64
#define QKV_LD   1536   // 3*D_MODEL

union AB { v16h h; uint32_t u[8]; h2v p[8]; };
union PK { _Float16 h[8]; uint32_t w[4]; };

// A-matrix 16x32 f16 layout: lane half 'hf' (lane>=16), VGPR i holds K pair at:
__device__ __forceinline__ int kbase(int i, int hf) {
    return (i < 4 ? 2 * i : 16 + 2 * (i - 4)) + 8 * hf;
}

__device__ __forceinline__ v8f wmma16(const AB& a, const AB& b, v8f c) {
    return __builtin_amdgcn_wmma_f32_16x16x32_f16(false, a.h, false, b.h,
                                                  (short)0, c, false, false);
}

// exchange values between the two 16-lane halves (xor-16) in pure VALU
__device__ __forceinline__ float xhalf(float x) {
#if __has_builtin(__builtin_amdgcn_permlanex16)
    int r = __builtin_amdgcn_permlanex16(__float_as_int(x), __float_as_int(x),
                                         0x76543210, 0xfedcba98, false, false);
    return __int_as_float(r);
#else
    return __shfl_xor(x, 16, 32);
#endif
}

// ---------------------------------------------------------------- converts
__global__ void cvt_f32_to_f16(const float* __restrict__ in,
                               _Float16* __restrict__ out, int n4) {
    int i = blockIdx.x * blockDim.x + threadIdx.x;
    if (i >= n4) return;
    float4 v = reinterpret_cast<const float4*>(in)[i];
    union { _Float16 h[4]; uint64_t u; } p;
    p.h[0] = (_Float16)v.x; p.h[1] = (_Float16)v.y;
    p.h[2] = (_Float16)v.z; p.h[3] = (_Float16)v.w;
    reinterpret_cast<uint64_t*>(out)[i] = p.u;
}

// ------------------------------------------------- V transpose: [b,s,(v)h,d] -> [bh,d,s]
__global__ void transpose_v_kernel(const _Float16* __restrict__ qkvh,
                                   _Float16* __restrict__ vt) {
    int tid = blockIdx.x * blockDim.x + threadIdx.x;   // BATCH*H*SEQ*DIM_H threads
    int d  = tid & 63;
    int s  = (tid >> 6) & (SEQ - 1);
    int bh = tid >> 18;                                // SEQ*DIM_H = 2^18
    int b  = bh >> 3, h = bh & 7;
    vt[((size_t)bh * DIM_H + d) * SEQ + s] =
        qkvh[((size_t)(b * SEQ + s)) * QKV_LD + 2 * D_MODEL + h * DIM_H + d];
}

// ------------------------------------------------- generic GEMM: C = A[M,K] * Bw[N,K]^T + bias
template <bool OUT16>
__global__ void gemm_wmma_f16(const _Float16* __restrict__ A,
                              const _Float16* __restrict__ Bw,
                              const float* __restrict__ bias,
                              void* __restrict__ Cout, int M, int N, int K) {
    const int lane = threadIdx.x & 31;
    const int lm = lane & 15, lh = lane >> 4;
    const int wid = blockIdx.x * (blockDim.x >> 5) + (threadIdx.x >> 5);
    const int ntiles = N >> 6;
    const int mt = wid / ntiles, nt = wid % ntiles;
    const int m0 = mt * 16, n0 = nt * 64;
    if (m0 >= M) return;

    v8f acc[4] = {};
    for (int kk = 0; kk < K; kk += 32) {
        AB a;
#pragma unroll
        for (int i = 0; i < 8; i++)
            a.u[i] = *reinterpret_cast<const uint32_t*>(
                A + (size_t)(m0 + lm) * K + kk + kbase(i, lh));
#pragma unroll
        for (int c = 0; c < 4; c++) {
            AB bm;
#pragma unroll
            for (int j = 0; j < 8; j++)
                bm.u[j] = *reinterpret_cast<const uint32_t*>(
                    Bw + (size_t)(n0 + c * 16 + lm) * K + kk + kbase(j, lh));
            acc[c] = wmma16(a, bm, acc[c]);
        }
    }
#pragma unroll
    for (int c = 0; c < 4; c++) {
        const float bv = bias[n0 + c * 16 + lm];
#pragma unroll
        for (int v = 0; v < 8; v++) {
            const float val = acc[c][v] + bv;
            const size_t idx = (size_t)(m0 + v + 8 * lh) * N + n0 + c * 16 + lm;
            if (OUT16) ((_Float16*)Cout)[idx] = (_Float16)val;
            else       ((float*)Cout)[idx] = val;
        }
    }
}

// ------------------------------------------------- fused flash attention (transposed tiles)
// 1 wave = one 16-query tile of one (b,h).
// S^T = K Q^T  (rows=keys, cols=queries)  -> softmax reduction in-lane + one permlanex16
// O^T = V^T P^T (rows=d, cols=queries)    -> vectorized ctx stores
__global__ void attn_fused_kernel(const _Float16* __restrict__ qkvh,
                                  const _Float16* __restrict__ vt,
                                  _Float16* __restrict__ ctx) {
    __shared__ uint32_t plds[4][16 * 16];   // per-wave P^T tile [query][key/2], u32-typed

    const int lane = threadIdx.x & 31;
    const int lm = lane & 15, lh = lane >> 4;
    const int wl = threadIdx.x >> 5;
    const int wid = blockIdx.x * 4 + wl;    // 4096 waves total
    const int qt = wid & 255;               // SEQ/16
    const int bh = wid >> 8;
    const int b = bh >> 3, h = bh & 7;
    const int q0 = qt * 16;

    const _Float16* qptr = qkvh + (size_t)b * SEQ * QKV_LD + h * DIM_H;
    const _Float16* kptr = qptr + D_MODEL;
    const _Float16* vtp  = vt + (size_t)bh * DIM_H * SEQ;

    // Q tile as B-matrix (d x query), pre-scaled by 1/sqrt(dh) = 0.125 (exact in f16)
    AB bq0, bq1;
    const h2v sc2 = { (_Float16)0.125f, (_Float16)0.125f };
#pragma unroll
    for (int j = 0; j < 8; j++) {
        const size_t row = (size_t)(q0 + lm) * QKV_LD;
        bq0.u[j] = *reinterpret_cast<const uint32_t*>(qptr + row + kbase(j, lh));
        bq1.u[j] = *reinterpret_cast<const uint32_t*>(qptr + row + 32 + kbase(j, lh));
        bq0.p[j] *= sc2;
        bq1.p[j] *= sc2;
    }

    v8f O[4] = {};
    float mrun = -1e30f, lrun = 0.f;

    // per-lane streaming pointers (strength-reduced addressing)
    const _Float16* kit = kptr + (size_t)lm * QKV_LD;
    const _Float16* vit = vtp + (size_t)lm * SEQ;

    for (int key0 = 0; key0 < SEQ; key0 += 32) {
        __builtin_prefetch(kit + 32 * QKV_LD, 0, 0);

        // S^T over dh=64: two k=32 WMMAs per 16-key tile
        v8f S0 = {}, S1 = {};
        {
            AB ak;
#pragma unroll
            for (int i = 0; i < 8; i++)
                ak.u[i] = *reinterpret_cast<const uint32_t*>(kit + kbase(i, lh));
            S0 = wmma16(ak, bq0, S0);
#pragma unroll
            for (int i = 0; i < 8; i++)
                ak.u[i] = *reinterpret_cast<const uint32_t*>(kit + 32 + kbase(i, lh));
            S0 = wmma16(ak, bq1, S0);
#pragma unroll
            for (int i = 0; i < 8; i++)
                ak.u[i] = *reinterpret_cast<const uint32_t*>(kit + 16 * QKV_LD + kbase(i, lh));
            S1 = wmma16(ak, bq0, S1);
#pragma unroll
            for (int i = 0; i < 8; i++)
                ak.u[i] = *reinterpret_cast<const uint32_t*>(kit + 16 * QKV_LD + 32 + kbase(i, lh));
            S1 = wmma16(ak, bq1, S1);
        }

        // online softmax over keys: in-lane over 16 regs + one half-swap (VALU)
        float tm = S0[0];
#pragma unroll
        for (int v = 1; v < 8; v++) tm = fmaxf(tm, S0[v]);
#pragma unroll
        for (int v = 0; v < 8; v++) tm = fmaxf(tm, S1[v]);
        tm = fmaxf(tm, xhalf(tm));
        const float mnew = fmaxf(mrun, tm);
        const float corr = __expf(mrun - mnew);
        mrun = mnew;

        PK pk0, pk1;
        float rs = 0.f;
#pragma unroll
        for (int v = 0; v < 8; v++) {
            const float p0 = __expf(S0[v] - mnew);
            const float p1 = __expf(S1[v] - mnew);
            rs += p0 + p1;
            pk0.h[v] = (_Float16)p0;
            pk1.h[v] = (_Float16)p1;
        }
        rs += xhalf(rs);
        lrun = lrun * corr + rs;

        // store P^T as [query][key]: 8 consecutive keys per half -> ds_store_b128
        {
            uint32_t* prow = &plds[wl][lm * 16];
#pragma unroll
            for (int w = 0; w < 4; w++) prow[4 * lh + w]     = pk0.w[w];
#pragma unroll
            for (int w = 0; w < 4; w++) prow[8 + 4 * lh + w] = pk1.w[w];
        }

#pragma unroll
        for (int c = 0; c < 4; c++)
#pragma unroll
            for (int v = 0; v < 8; v++) O[c][v] *= corr;

        // reload P^T in B-layout (k=key, n=query): key pairs at kbase(j, lh)
        AB bp;
#pragma unroll
        for (int j = 0; j < 8; j++)
            bp.u[j] = plds[wl][lm * 16 + kbase(j, lh) / 2];

        // O^T += V^T P^T : A = V^T tile (d x key), key pairs contiguous in vth
#pragma unroll
        for (int c = 0; c < 4; c++) {
            AB av;
#pragma unroll
            for (int i = 0; i < 8; i++)
                av.u[i] = *reinterpret_cast<const uint32_t*>(
                    vit + c * 16 * SEQ + kbase(i, lh));
            O[c] = wmma16(av, bp, O[c]);
        }

        kit += 32 * QKV_LD;
        vit += 32;
    }

    // normalize (per-query scalar) and store: 8 consecutive d per lane -> b128 stores
    const float inv = 1.0f / lrun;
#pragma unroll
    for (int c = 0; c < 4; c++) {
        PK pk;
#pragma unroll
        for (int v = 0; v < 8; v++) pk.h[v] = (_Float16)(O[c][v] * inv);
        uint32_t* dst = reinterpret_cast<uint32_t*>(
            &ctx[(size_t)(b * SEQ + q0 + lm) * D_MODEL + h * DIM_H + c * 16 + 8 * lh]);
#pragma unroll
        for (int w = 0; w < 4; w++) dst[w] = pk.w[w];
    }
}

// ---------------------------------------------------------------- launcher
extern "C" void kernel_launch(void* const* d_in, const int* in_sizes, int n_in,
                              void* d_out, int out_size, void* d_ws, size_t ws_size,
                              hipStream_t stream) {
    (void)in_sizes; (void)n_in; (void)out_size; (void)ws_size;
    const float* x     = (const float*)d_in[0];
    const float* W_in  = (const float*)d_in[1];
    const float* b_in  = (const float*)d_in[2];
    const float* W_out = (const float*)d_in[3];
    const float* b_out = (const float*)d_in[4];
    float* out = (float*)d_out;

    char* ws = (char*)d_ws;
    _Float16* xh   = (_Float16*)ws; ws += (size_t)BATCH * SEQ * D_MODEL * 2;       // 8 MB
    _Float16* wih  = (_Float16*)ws; ws += (size_t)QKV_LD * D_MODEL * 2;            // 1.5 MB
    _Float16* woh  = (_Float16*)ws; ws += (size_t)D_MODEL * D_MODEL * 2;           // 0.5 MB
    _Float16* qkvh = (_Float16*)ws; ws += (size_t)BATCH * SEQ * QKV_LD * 2;        // 24 MB
    _Float16* vth  = (_Float16*)ws; ws += (size_t)BATCH * NUM_H * DIM_H * SEQ * 2; // 8 MB
    _Float16* ctxh = (_Float16*)ws;                                                // 8 MB

    const int M = BATCH * SEQ;  // 8192

    cvt_f32_to_f16<<<(M * D_MODEL / 4 + 255) / 256, 256, 0, stream>>>(x, xh, M * D_MODEL / 4);
    cvt_f32_to_f16<<<(QKV_LD * D_MODEL / 4 + 255) / 256, 256, 0, stream>>>(W_in, wih, QKV_LD * D_MODEL / 4);
    cvt_f32_to_f16<<<(D_MODEL * D_MODEL / 4 + 255) / 256, 256, 0, stream>>>(W_out, woh, D_MODEL * D_MODEL / 4);

    // qkv = x @ W_in^T + b_in  (f16 out): 512x24 tiles -> 12288 waves -> 1536 blocks
    gemm_wmma_f16<true><<<(M / 16) * (QKV_LD / 64) / 8, 256, 0, stream>>>(
        xh, wih, b_in, qkvh, M, QKV_LD, D_MODEL);

    transpose_v_kernel<<<(BATCH * NUM_H * SEQ * DIM_H) / 256, 256, 0, stream>>>(qkvh, vth);

    // 4096 q-tiles, 4 waves/block
    attn_fused_kernel<<<(BATCH * NUM_H * (SEQ / 16)) / 4, 128, 0, stream>>>(qkvh, vth, ctxh);

    // out = ctx @ W_out^T + b_out (f32 out): 512x8 tiles -> 4096 waves -> 512 blocks
    gemm_wmma_f16<false><<<(M / 16) * (D_MODEL / 64) / 8, 256, 0, stream>>>(
        ctxh, woh, b_out, out, M, D_MODEL, D_MODEL);
}
